// NSA_86174223827252
// MI455X (gfx1250) — matrix operand order repaired and checked by reading, not compiled, not run on hardware
//
#include <hip/hip_runtime.h>

// ---------------------------------------------------------------------------
// NSA forward for MI455X (gfx1250, wave32, WMMA bf16 16x16x32, fp32 accum).
// Q/K/V are produced in bf16 by the projection GEMM epilogues (V transposed
// [dim][seq]) so all attention staging is pure u32 copies into LDS and WMMA
// operand builds are pure ds_load_b128 pairs. Next-tile global_prefetch_b8
// overlaps HBM/L2 latency with WMMA work.
// ---------------------------------------------------------------------------

#define S_LEN   2048
#define HEADS   16
#define HDIM    64
#define DIM     1024        // HEADS * HDIM
#define SCMP    128         // n_win + 1
#define NBLK    32          // S_LEN / SEL_BS
#define SCALE   0.125f      // 1/sqrt(64)

typedef __attribute__((ext_vector_type(16))) __bf16 v16bf;
typedef __attribute__((ext_vector_type(8)))  float  v8f;
typedef unsigned short u16;

// f32 -> bf16 bits (round-half-up)
static __device__ inline u16 f2bfbits(float x) {
    union { float f; unsigned u; } c; c.f = x;
    return (u16)((c.u + 0x8000u) >> 16);
}
// two f32 -> packed bf16x2: 2x v_add + 1x v_perm_b32
static __device__ inline unsigned packbf(float a, float b) {
    union { float f; unsigned u; } x, y;
    x.f = a; y.f = b;
    unsigned ra = x.u + 0x8000u;
    unsigned rb = y.u + 0x8000u;
    return __builtin_amdgcn_perm(rb, ra, 0x07060302u);
}
// bf16 bits -> f32
static __device__ inline float bf2f(u16 h) {
    union { unsigned u; float f; } c; c.u = ((unsigned)h) << 16;
    return c.f;
}

static __device__ inline v8f vzero8() {
    v8f z = {0.f,0.f,0.f,0.f,0.f,0.f,0.f,0.f};
    return z;
}

static __device__ inline v8f wmma32(v16bf a, v16bf b, v8f c) {
    return __builtin_amdgcn_wmma_f32_16x16x32_bf16(
        false, a, false, b, (short)0, c, false, false);
}

// A-operand (16x32 bf16) from a 16-u32 packed row. kb = (lane&16)?8:0.
static __device__ inline v16bf frag_a_pk(const unsigned* row16, int kb) {
    union { v16bf v; unsigned u[8]; } t;
    int jb = kb >> 1;
#pragma unroll
    for (int i2 = 0; i2 < 8; ++i2)
        t.u[i2] = row16[jb + (i2 < 4 ? i2 : i2 + 4)];
    return t.v;
}
// B-operand (32x16 bf16): kb2 = (lane&16)?16:0; u[i2] = row[kb2/4 + (i2 < 4 ? i2 : i2 + 4)]
static __device__ inline v16bf frag_b_pk(const unsigned* row, int kb2) {
    union { v16bf v; unsigned u[8]; } t;
    int jb = kb2 >> 2;
#pragma unroll
    for (int i2 = 0; i2 < 8; ++i2)
        t.u[i2] = row[jb + (i2 < 4 ? i2 : i2 + 4)];
    return t.v;
}
// A-operand packed on the fly from an f32 row (P tiles; single consumer)
static __device__ inline v16bf frag_a_packf32(const float* row, int kb) {
    union { v16bf v; unsigned u[8]; } t;
#pragma unroll
    for (int i2 = 0; i2 < 8; ++i2) {
        int kk = kb + (i2 < 4 ? 2 * i2 : 2 * i2 + 8);
        t.u[i2] = packbf(row[kk], row[kk + 1]);
    }
    return t.v;
}

// ---------------------------------------------------------------------------
// Generic GEMM: act( A[M,K]_f32 @ B[K,N]_f32 ) with bf16 WMMA.
// outfmt: 0 = f32 row-major, 1 = bf16 row-major, 2 = bf16 transposed [N][M].
// act: 0 none, 1 silu, 2 sigmoid.  M%64==0, K%32==0.
// ---------------------------------------------------------------------------
__global__ __launch_bounds__(128)
void nsa_gemm(const float* __restrict__ A, const float* __restrict__ B,
              void* __restrict__ C, int M, int N, int K, int act, int outfmt) {
    const int row0 = blockIdx.y * 64;
    const int col0 = blockIdx.x * 64;
    const int tid  = threadIdx.x;
    const int lane = tid & 31;
    const int wave = tid >> 5;
    const int kb   = (lane & 16) ? 8 : 0;
    const int kb2  = (lane & 16) ? 16 : 0;
    const int n    = lane & 15;

    __shared__ unsigned lAp[64][16];   // [row][k-pair]
    __shared__ unsigned lBTp[64][16];  // [col][k-pair] (transposed B)

    v8f acc[4];
#pragma unroll
    for (int nt = 0; nt < 4; ++nt) acc[nt] = vzero8();

    for (int k0 = 0; k0 < K; k0 += 32) {
        if (k0 + 32 < K) {   // prefetch next k-step (global_prefetch_b8)
            __builtin_prefetch(A + (size_t)(row0 + (tid & 63)) * K + k0 + 32, 0, 1);
            __builtin_prefetch(B + (size_t)(k0 + 32 + (tid & 31)) * N + col0 + ((tid >> 5) << 4), 0, 1);
        }
#pragma unroll
        for (int it = 0; it < 8; ++it) {
            int idx = tid + it * 128;
            int r = idx >> 4, j = idx & 15;
            const float* ga = A + (size_t)(row0 + r) * K + k0 + 2 * j;
            lAp[r][j] = packbf(ga[0], ga[1]);
        }
#pragma unroll
        for (int it = 0; it < 8; ++it) {
            int idx = tid + it * 128;
            int c = idx & 63, j = idx >> 6;
            int col = col0 + c;
            float b0 = 0.f, b1 = 0.f;
            if (col < N) {
                b0 = B[(size_t)(k0 + 2 * j) * N + col];
                b1 = B[(size_t)(k0 + 2 * j + 1) * N + col];
            }
            lBTp[c][j] = packbf(b0, b1);
        }
        __syncthreads();

        v16bf af = frag_a_pk(&lAp[wave * 16 + (lane & 15)][0], kb);
#pragma unroll
        for (int nt = 0; nt < 4; ++nt) {
            v16bf bf_ = frag_b_pk(&lBTp[nt * 16 + n][0], kb2);
            acc[nt] = wmma32(af, bf_, acc[nt]);
        }
        __syncthreads();
    }

    const int half = (lane & 16) ? 8 : 0;
#pragma unroll
    for (int nt = 0; nt < 4; ++nt) {
        int col = col0 + nt * 16 + n;
        if (col >= N) continue;
        float vals[8];
#pragma unroll
        for (int r = 0; r < 8; ++r) {
            float x = acc[nt][r];
            if (act == 1)      x = x / (1.f + __expf(-x));   // silu
            else if (act == 2) x = 1.f / (1.f + __expf(-x)); // sigmoid
            vals[r] = x;
        }
        int rowb = row0 + wave * 16 + half;
        if (outfmt == 0) {
            float* Cf = (float*)C;
#pragma unroll
            for (int r = 0; r < 8; ++r) Cf[(size_t)(rowb + r) * N + col] = vals[r];
        } else if (outfmt == 1) {
            u16* Ch = (u16*)C;
#pragma unroll
            for (int r = 0; r < 8; ++r) Ch[(size_t)(rowb + r) * N + col] = f2bfbits(vals[r]);
        } else {            // bf16 transposed [N][M]; rows consecutive -> u32 stores
            unsigned* Cu = (unsigned*)C;
            size_t base = ((size_t)col * M + rowb) >> 1;
#pragma unroll
            for (int i = 0; i < 4; ++i)
                Cu[base + i] = packbf(vals[2 * i], vals[2 * i + 1]);
        }
    }
}

// ---------------------------------------------------------------------------
// Compression: inputs K (bf16 [S][DIM]) and V (bf16 transposed [DIM][S]).
// Outputs: k_cmp bf16 [H][128][64], v_cmp bf16 transposed [H][64][128].
// ---------------------------------------------------------------------------
__global__ __launch_bounds__(64)
void nsa_compress(const u16* __restrict__ kin, const u16* __restrict__ vinT,
                  const float* __restrict__ pe_k, const float* __restrict__ down_k,
                  const float* __restrict__ stop_k,
                  const float* __restrict__ pe_v, const float* __restrict__ down_v,
                  const float* __restrict__ stop_v,
                  u16* __restrict__ kcmp, u16* __restrict__ vcmpT) {
    const int w     = blockIdx.x;           // 0..127 output slot
    const int h     = blockIdx.y;
    const int which = blockIdx.z;           // 0 = K, 1 = V
    const int d     = threadIdx.x;          // 0..63

    const float* pe   = which ? pe_v   : pe_k;
    const float* down = which ? down_v : down_k;
    const float* stop = which ? stop_v : stop_k;

    if (w == 0) {
        if (which == 0) kcmp[((size_t)h * SCMP + 0) * HDIM + d] = 0;
        else            vcmpT[((size_t)h * HDIM + d) * SCMP + 0] = 0;
        return;
    }
    const int win = w - 1;

    __shared__ float bufA[32][64];
    __shared__ float bufB[16][64];

    for (int idx = d; idx < 32 * 64; idx += 64) {
        int i = idx >> 6, c = idx & 63;
        float x = which
            ? bf2f(vinT[((size_t)h * HDIM + c) * S_LEN + win * 16 + i])
            : bf2f(kin[(size_t)(win * 16 + i) * DIM + h * HDIM + c]);
        bufA[i][c] = x + pe[i * 64 + c];
    }
    __syncthreads();

    int rout = 16;
#pragma unroll
    for (int l = 0; l < 5; ++l) {
        const float* dl = down + l * 128 * 64;
        for (int r = 0; r < rout; ++r) {
            float a = 0.f;
            for (int e = 0; e < 64; ++e) a += bufA[2 * r][e]     * dl[e * 64 + d];
            for (int e = 0; e < 64; ++e) a += bufA[2 * r + 1][e] * dl[(64 + e) * 64 + d];
            bufB[r][d] = a / (1.f + __expf(-a));                 // silu
        }
        __syncthreads();
        for (int r = 0; r < rout; ++r) bufA[r][d] = bufB[r][d];
        __syncthreads();
        rout >>= 1;
    }

    float a = 0.f;
    for (int e = 0; e < 64; ++e) a += bufA[0][e] * stop[e * 64 + d];
    if (which == 0) kcmp[((size_t)h * SCMP + w) * HDIM + d] = f2bfbits(a);
    else            vcmpT[((size_t)h * HDIM + d) * SCMP + w] = f2bfbits(a);
}

// ---------------------------------------------------------------------------
// Compressed-branch attention + conv1d block importance.
// All staging = pure u32 copies of packed bf16.
// ---------------------------------------------------------------------------
__global__ __launch_bounds__(32)
void nsa_attn_cmp(const u16* __restrict__ qbf, const u16* __restrict__ kc,
                  const u16* __restrict__ vcT,
                  float* __restrict__ o_cmp, float* __restrict__ p_slc) {
    const int t    = blockIdx.x;            // 0..127
    const int h    = blockIdx.y;
    const int lane = threadIdx.x;
    const int half = (lane & 16) ? 8 : 0;
    const int n    = lane & 15;
    const int kb   = (lane & 16) ? 8 : 0;
    const int kb2  = (lane & 16) ? 16 : 0;

    __shared__ unsigned lqp[16][32];        // Q packed
    __shared__ unsigned lkcp[SCMP][32];     // k_cmp packed [key][d-pair]
    __shared__ unsigned lvcTp[64][64];      // v_cmp packed [d][key-pair]
    __shared__ float    lsp[16][SCMP];      // scores -> probs

    const unsigned* qu = (const unsigned*)qbf;
    const unsigned* ku = (const unsigned*)kc;
    const unsigned* vu = (const unsigned*)vcT;

#pragma unroll
    for (int it = 0; it < 16; ++it) {
        int idx = lane + it * 32;
        int r = idx >> 5, j = idx & 31;
        lqp[r][j] = qu[(((size_t)(t * 16 + r) * DIM + h * HDIM) >> 1) + j];
    }
#pragma unroll 8
    for (int it = 0; it < 128; ++it) {
        int idx = lane + it * 32;
        int c = idx >> 5, j = idx & 31;
        lkcp[c][j] = ku[((size_t)h * SCMP + c) * 32 + j];
    }
#pragma unroll 8
    for (int it = 0; it < 128; ++it) {
        int idx = lane + it * 32;
        int d = idx & 63, j = idx >> 6;
        lvcTp[d][j] = vu[((size_t)h * HDIM + d) * 64 + j];
    }
    __syncthreads();

    v16bf qa0 = frag_a_pk(&lqp[lane & 15][0],  kb);
    v16bf qa1 = frag_a_pk(&lqp[lane & 15][16], kb);

    // scores over all 128 compressed keys (8 key tiles of 16)
    for (int ct = 0; ct < 8; ++ct) {
        v8f s = vzero8();
        s = wmma32(qa0, frag_b_pk(&lkcp[ct * 16 + n][0],  kb2), s);
        s = wmma32(qa1, frag_b_pk(&lkcp[ct * 16 + n][16], kb2), s);
#pragma unroll
        for (int r = 0; r < 8; ++r) {
            int row = r + half, qi = t * 16 + row;
            int rr  = qi >> 4;                       // query's cmp row
            int c   = ct * 16 + n;
            bool ok = (c < rr) || (c == 0 && rr < 32);
            lsp[row][c] = ok ? s[r] * SCALE : -1e30f;
        }
    }
    __syncthreads();

    // softmax: 2 lanes per row (each handles 64 cols)
    {
        int row  = lane >> 1;
        int base = (lane & 1) * 64;
        float mx = -1e30f;
        for (int c = 0; c < 64; ++c) mx = fmaxf(mx, lsp[row][base + c]);
        mx = fmaxf(mx, __shfl_xor(mx, 1, 32));
        float sm = 0.f;
        for (int c = 0; c < 64; ++c) {
            float p = __expf(lsp[row][base + c] - mx);
            lsp[row][base + c] = p;
            sm += p;
        }
        sm += __shfl_xor(sm, 1, 32);
        float inv = (sm > 0.f) ? 1.f / sm : 0.f;
        for (int c = 0; c < 64; ++c) lsp[row][base + c] *= inv;
    }
    __syncthreads();

    // conv1d [1,2,2,2,1], stride 4, pad 1 -> 32 block scores per row
    {
        int row = lane >> 1, qi = t * 16 + row;
        int nb0 = (lane & 1) * 16;
        const float kw[5] = {1.f, 2.f, 2.f, 2.f, 1.f};
        for (int j = 0; j < 16; ++j) {
            int nb = nb0 + j;
            float a = 0.f;
#pragma unroll
            for (int tt = 0; tt < 5; ++tt) {
                int ci = 4 * nb - 1 + tt;
                if (ci >= 0 && ci < SCMP) a += kw[tt] * lsp[row][ci];
            }
            p_slc[((size_t)h * S_LEN + qi) * NBLK + nb] = a;
        }
    }

    // P @ v_cmp (P normalized in lsp)
    v8f oc[4];
#pragma unroll
    for (int nt = 0; nt < 4; ++nt) oc[nt] = vzero8();
    for (int st = 0; st < 4; ++st) {
        v16bf pa = frag_a_packf32(&lsp[lane & 15][32 * st], kb);
#pragma unroll
        for (int nt = 0; nt < 4; ++nt)
            oc[nt] = wmma32(pa, frag_b_pk(&lvcTp[nt * 16 + n][16 * st], kb2), oc[nt]);
    }
#pragma unroll
    for (int nt = 0; nt < 4; ++nt)
#pragma unroll
        for (int r = 0; r < 8; ++r) {
            int row = r + half, qi = t * 16 + row;
            o_cmp[(size_t)qi * DIM + h * HDIM + nt * 16 + n] = oc[nt][r];
        }
}

// ---------------------------------------------------------------------------
// Top-16 block selection (jax.lax.top_k semantics: ties -> lower index).
// ---------------------------------------------------------------------------
__global__ void nsa_topk(const float* __restrict__ p_slc, unsigned* __restrict__ blkmask) {
    int idx = blockIdx.x * blockDim.x + threadIdx.x;
    if (idx >= HEADS * S_LEN) return;
    int h = idx >> 11, s = idx & (S_LEN - 1);
    const float* p = p_slc + ((size_t)h * S_LEN + s) * NBLK;
    float v[NBLK];
#pragma unroll
    for (int j = 0; j < NBLK; ++j) v[j] = p[j];
    unsigned chosen = 0;
    for (int it = 0; it < 16; ++it) {
        float best = -3.0e38f; int bi = 0;
        for (int j = 0; j < NBLK; ++j)
            if (!((chosen >> j) & 1u) && v[j] > best) { best = v[j]; bi = j; }
        chosen |= 1u << bi;
    }
    int qb = s >> 6;
    unsigned allowed = (qb >= 32) ? 0xffffffffu : ((1u << qb) - 1u);
    blkmask[(size_t)h * S_LEN + s] = chosen & allowed;
}

// ---------------------------------------------------------------------------
// Full attention: selected + sliding-window branches share WMMA scores.
// Flash-style, one wave32 per (q-tile, head); staging = u32 copies; next
// super-tile prefetched (global_prefetch_b8) while WMMAs run.
// ---------------------------------------------------------------------------
__global__ __launch_bounds__(32)
void nsa_attn_full(const u16* __restrict__ qbf, const u16* __restrict__ kbf,
                   const u16* __restrict__ vTbf, const unsigned* __restrict__ blkmask,
                   float* __restrict__ o_slc, float* __restrict__ o_swa) {
    const int t    = blockIdx.x;            // q tile 0..127
    const int h    = blockIdx.y;
    const int lane = threadIdx.x;
    const int half = (lane & 16) ? 8 : 0;
    const int n    = lane & 15;
    const int kb   = (lane & 16) ? 8 : 0;
    const int kb2  = (lane & 16) ? 16 : 0;

    __shared__ unsigned lqp[16][32];   // Q packed
    __shared__ unsigned lkp[32][32];   // K packed: [key][d-pair]
    __shared__ unsigned lvTp[64][16];  // V packed transposed: [d][key-pair]
    __shared__ float    lp[16][32];    // probabilities (f32)
    __shared__ unsigned lbm[16];

    const unsigned* qu = (const unsigned*)qbf;
    const unsigned* ku = (const unsigned*)kbf;
    const unsigned* vu = (const unsigned*)vTbf;

#pragma unroll
    for (int it = 0; it < 16; ++it) {
        int idx = lane + it * 32;
        int r = idx >> 5, j = idx & 31;
        lqp[r][j] = qu[(((size_t)(t * 16 + r) * DIM + h * HDIM) >> 1) + j];
    }
    if (lane < 16) lbm[lane] = blkmask[(size_t)h * S_LEN + t * 16 + lane];
    __syncthreads();

    v16bf qa0 = frag_a_pk(&lqp[lane & 15][0],  kb);
    v16bf qa1 = frag_a_pk(&lqp[lane & 15][16], kb);

    v8f oS[4], oW[4];
    float mS[8], lS[8], mW[8], lW[8];
#pragma unroll
    for (int nt = 0; nt < 4; ++nt) { oS[nt] = vzero8(); oW[nt] = vzero8(); }
#pragma unroll
    for (int r = 0; r < 8; ++r) { mS[r] = -1e30f; lS[r] = 0.f; mW[r] = -1e30f; lW[r] = 0.f; }

    const int nsup = (t >> 1) + 1;          // 32-key super tiles cover keys <= 16t+15
    for (int jt = 0; jt < nsup; ++jt) {
        const int k0 = jt * 32;

        if (jt + 1 < nsup) {                 // prefetch next super-tile
            __builtin_prefetch(kbf + (size_t)(k0 + 32 + lane) * DIM + h * HDIM, 0, 1);
            __builtin_prefetch(vTbf + ((size_t)h * HDIM + lane) * S_LEN + k0 + 32, 0, 1);
            __builtin_prefetch(vTbf + ((size_t)h * HDIM + 32 + lane) * S_LEN + k0 + 32, 0, 1);
        }
#pragma unroll
        for (int it = 0; it < 32; ++it) {
            int idx = lane + it * 32;
            int key = idx >> 5, j = idx & 31;
            lkp[key][j] = ku[(((size_t)(k0 + key) * DIM + h * HDIM) >> 1) + j];
        }
#pragma unroll
        for (int it = 0; it < 32; ++it) {
            int idx = lane + it * 32;
            int d = idx & 63, j = idx >> 6;
            lvTp[d][j] = vu[((size_t)h * HDIM + d) * (S_LEN / 2) + (k0 >> 1) + j];
        }
        __syncthreads();

        // scores: 16 queries x 32 keys (two 16x16 C tiles)
        v8f s0 = vzero8(), s1 = vzero8();
        s0 = wmma32(qa0, frag_b_pk(&lkp[n][0],       kb2), s0);
        s1 = wmma32(qa0, frag_b_pk(&lkp[16 + n][0],  kb2), s1);
        s0 = wmma32(qa1, frag_b_pk(&lkp[n][16],      kb2), s0);
        s1 = wmma32(qa1, frag_b_pk(&lkp[16 + n][16], kb2), s1);

        // V B-fragments (shared by both branches)
        v16bf vb[4];
#pragma unroll
        for (int nt = 0; nt < 4; ++nt)
            vb[nt] = frag_b_pk(&lvTp[nt * 16 + n][0], kb2);

        float sv[8][2];
#pragma unroll
        for (int r = 0; r < 8; ++r) { sv[r][0] = s0[r] * SCALE; sv[r][1] = s1[r] * SCALE; }

        for (int br = 0; br < 2; ++br) {
            float* rm   = br ? mW : mS;
            float* rl   = br ? lW : lS;
            v8f*   oacc = br ? oW : oS;

            __syncthreads();                 // lp reuse fence
#pragma unroll
            for (int r = 0; r < 8; ++r) {
                int row = r + half, qi = t * 16 + row;
                int kj0 = k0 + n, kj1 = k0 + 16 + n;
                bool ok0, ok1;
                if (br == 0) {               // selected blocks (mask includes "allowed")
                    unsigned bm = lbm[row];
                    ok0 = (bm >> (kj0 >> 6)) & 1u;
                    ok1 = (bm >> (kj1 >> 6)) & 1u;
                } else {                     // sliding window
                    ok0 = (kj0 <= qi) && (qi - kj0 <= 512);
                    ok1 = (kj1 <= qi) && (qi - kj1 <= 512);
                }
                float a0 = ok0 ? sv[r][0] : -1e30f;
                float a1 = ok1 ? sv[r][1] : -1e30f;
                float tmax = fmaxf(a0, a1);
                for (int sft = 1; sft < 16; sft <<= 1)
                    tmax = fmaxf(tmax, __shfl_xor(tmax, sft, 32));
                float nm   = fmaxf(rm[r], tmax);
                float corr = __expf(rm[r] - nm);
                float p0 = ok0 ? __expf(sv[r][0] - nm) : 0.f;
                float p1 = ok1 ? __expf(sv[r][1] - nm) : 0.f;
                float ts = p0 + p1;
                for (int sft = 1; sft < 16; sft <<= 1)
                    ts += __shfl_xor(ts, sft, 32);
                rl[r] = rl[r] * corr + ts;
                rm[r] = nm;
#pragma unroll
                for (int nt = 0; nt < 4; ++nt) oacc[nt][r] *= corr;
                lp[row][n]      = p0;
                lp[row][16 + n] = p1;
            }
            __syncthreads();

            v16bf pa = frag_a_packf32(&lp[lane & 15][0], kb);
#pragma unroll
            for (int nt = 0; nt < 4; ++nt)
                oacc[nt] = wmma32(pa, vb[nt], oacc[nt]);
        }
        __syncthreads();                     // before lkp/lvTp overwrite
    }

#pragma unroll
    for (int nt = 0; nt < 4; ++nt)
#pragma unroll
        for (int r = 0; r < 8; ++r) {
            int row = r + half, qi = t * 16 + row;
            size_t off = (size_t)qi * DIM + (size_t)h * HDIM + nt * 16 + n;
            o_slc[off] = (lS[r] > 0.f) ? oS[nt][r] / lS[r] : 0.f;
            o_swa[off] = (lW[r] > 0.f) ? oW[nt][r] / lW[r] : 0.f;
        }
}

// ---------------------------------------------------------------------------
// Gated combine: o = o_cmp*g0 + o_slc*g1 + o_swa*g2
// ---------------------------------------------------------------------------
__global__ void nsa_combine(const float* __restrict__ ocmp, const float* __restrict__ oslc,
                            const float* __restrict__ oswa, const float* __restrict__ g,
                            float* __restrict__ ocomb) {
    size_t idx = (size_t)blockIdx.x * blockDim.x + threadIdx.x;
    if (idx >= (size_t)S_LEN * DIM) return;
    int s = (int)(idx >> 10), hd = (int)(idx & 1023), h = hd >> 6;
    const float* gr = g + (size_t)s * (3 * HEADS) + h * 3;
    ocomb[idx] = ocmp[idx] * gr[0] + oslc[idx] * gr[1] + oswa[idx] * gr[2];
}

// ---------------------------------------------------------------------------
// Host launch
// ---------------------------------------------------------------------------
extern "C" void kernel_launch(void* const* d_in, const int* in_sizes, int n_in,
                              void* d_out, int out_size, void* d_ws, size_t ws_size,
                              hipStream_t stream) {
    (void)in_sizes; (void)n_in; (void)out_size; (void)ws_size;

    const float* x      = (const float*)d_in[0];
    const float* wq     = (const float*)d_in[1];
    const float* wk     = (const float*)d_in[2];
    const float* wv     = (const float*)d_in[3];
    const float* wo     = (const float*)d_in[4];
    const float* gw1    = (const float*)d_in[5];
    const float* gw2    = (const float*)d_in[6];
    const float* pe_k   = (const float*)d_in[7];
    const float* down_k = (const float*)d_in[8];
    const float* stop_k = (const float*)d_in[9];
    const float* pe_v   = (const float*)d_in[10];
    const float* down_v = (const float*)d_in[11];
    const float* stop_v = (const float*)d_in[12];

    float* W = (float*)d_ws;
    auto alloc = [&](size_t nfloats) { float* p = W; W += nfloats; return p; };
    u16*      qbf   = (u16*)alloc((size_t)S_LEN * DIM / 2);       // bf16 [S][DIM]
    u16*      kbf   = (u16*)alloc((size_t)S_LEN * DIM / 2);       // bf16 [S][DIM]
    u16*      vTbf  = (u16*)alloc((size_t)S_LEN * DIM / 2);       // bf16 [DIM][S]
    float*    t1    = alloc((size_t)S_LEN * 512);
    float*    gbuf  = alloc((size_t)S_LEN * 48);
    u16*      kcmp  = (u16*)alloc((size_t)HEADS * SCMP * HDIM / 2); // bf16 [H][128][64]
    u16*      vcmpT = (u16*)alloc((size_t)HEADS * SCMP * HDIM / 2); // bf16 [H][64][128]
    float*    pslc  = alloc((size_t)HEADS * S_LEN * NBLK);
    unsigned* bm    = (unsigned*)alloc((size_t)HEADS * S_LEN);
    float*    ocmp  = alloc((size_t)S_LEN * DIM);
    float*    oslc  = alloc((size_t)S_LEN * DIM);
    float*    oswa  = alloc((size_t)S_LEN * DIM);
    float*    ocomb = alloc((size_t)S_LEN * DIM);

    // Projections (WMMA GEMMs) -> bf16 outputs; V transposed [DIM][S]
    nsa_gemm<<<dim3(16, 32), 128, 0, stream>>>(x, wq, qbf,  S_LEN, DIM, DIM, 0, 1);
    nsa_gemm<<<dim3(16, 32), 128, 0, stream>>>(x, wk, kbf,  S_LEN, DIM, DIM, 0, 1);
    nsa_gemm<<<dim3(16, 32), 128, 0, stream>>>(x, wv, vTbf, S_LEN, DIM, DIM, 0, 2);

    // Gate MLP (f32 outputs)
    nsa_gemm<<<dim3(8, 32),  128, 0, stream>>>(x,  gw1, t1,   S_LEN, 512, DIM, 1, 0);
    nsa_gemm<<<dim3(1, 32),  128, 0, stream>>>(t1, gw2, gbuf, S_LEN, 48,  512, 2, 0);

    // K/V compression (bf16 in, bf16 out; V sides transposed)
    nsa_compress<<<dim3(SCMP, HEADS, 2), 64, 0, stream>>>(
        kbf, vTbf, pe_k, down_k, stop_k, pe_v, down_v, stop_v, kcmp, vcmpT);

    // Compressed-branch attention + block importance
    nsa_attn_cmp<<<dim3(S_LEN / 16, HEADS), 32, 0, stream>>>(qbf, kcmp, vcmpT, ocmp, pslc);

    // Top-16 block selection
    nsa_topk<<<(HEADS * S_LEN + 255) / 256, 256, 0, stream>>>(pslc, bm);

    // Selected + sliding-window attention (shared scores, flash-style)
    nsa_attn_full<<<dim3(S_LEN / 16, HEADS), 32, 0, stream>>>(qbf, kbf, vTbf, bm, oslc, oswa);

    // Gated combine + output projection
    nsa_combine<<<(int)(((size_t)S_LEN * DIM + 255) / 256), 256, 0, stream>>>(
        ocmp, oslc, oswa, gbuf, ocomb);
    nsa_gemm<<<dim3(16, 32), 128, 0, stream>>>(ocomb, wo, (float*)d_out, S_LEN, DIM, DIM, 0, 0);
}
